// PointViewFusion_43173011259722
// MI455X (gfx1250) — compile-verified, matrix operand order checked by reading, not run on hardware
//
#include <hip/hip_runtime.h>
#include <hip/hip_bf16.h>
#include <math.h>

// ---------------- problem constants ----------------
#define B_   4
#define M_   8
#define NP_  4096
#define PD_  256
#define VD_  256
#define H_   256
#define FD_  512
#define S_   128

typedef _Float16 v16h __attribute__((ext_vector_type(16)));
typedef _Float16 v8h  __attribute__((ext_vector_type(8)));
typedef float    v8f  __attribute__((ext_vector_type(8)));
typedef float    v4f  __attribute__((ext_vector_type(4)));

// ---------------- WMMA helpers ----------------
__device__ __forceinline__ v8f wmma_f16(v16h a, v16h b, v8f c) {
  // D = A(16x32 f16) x B(32x16 f16) + C(16x16 f32)
  return __builtin_amdgcn_wmma_f32_16x16x32_f16(
      /*neg_a=*/false, a, /*neg_b=*/false, b,
      /*c_mod=*/(short)0, c, /*reuse_a=*/false, /*reuse_b=*/false);
}

// A fragment (16x32) loaded from row-major f32 matrix, converted to f16.
// Lane layout: row = lane&15; lanes<16 hold K = k0+{0..7, 16..23},
// lanes>=16 hold K = k0+{8..15, 24..31}.
__device__ __forceinline__ v16h load_a_g(const float* __restrict__ A, int lda,
                                         int k0, int lane) {
  const int row = lane & 15;
  const int kb  = k0 + ((lane < 16) ? 0 : 8);
  const float* p = A + (size_t)row * lda + kb;
  v4f x0 = *(const v4f*)(p + 0);
  v4f x1 = *(const v4f*)(p + 4);
  v4f x2 = *(const v4f*)(p + 16);
  v4f x3 = *(const v4f*)(p + 20);
  v16h a;
#pragma unroll
  for (int i = 0; i < 4; ++i) {
    a[i]      = (_Float16)x0[i];
    a[4 + i]  = (_Float16)x1[i];
    a[8 + i]  = (_Float16)x2[i];
    a[12 + i] = (_Float16)x3[i];
  }
  return a;
}

// A fragment from row-major f16 matrix in LDS.
__device__ __forceinline__ v16h load_a_s(const _Float16* __restrict__ A, int lda,
                                         int k0, int lane) {
  const int row = lane & 15;
  const int kb  = k0 + ((lane < 16) ? 0 : 8);
  const _Float16* p = A + (size_t)row * lda + kb;
  v8h lo = *(const v8h*)(p);
  v8h hi = *(const v8h*)(p + 16);
  v16h a;
#pragma unroll
  for (int i = 0; i < 8; ++i) { a[i] = lo[i]; a[8 + i] = hi[i]; }
  return a;
}

// B fragment from prepacked weights: layout [(kt*NT + nt)*32 + lane][16 halves].
__device__ __forceinline__ v16h load_b(const _Float16* __restrict__ pack, int NT,
                                       int kt, int nt, int lane) {
  return *(const v16h*)(pack + (((size_t)kt * NT + nt) * 32 + lane) * 16);
}

// Stage a contiguous 16KB block (16 rows x 256 f32) from global into LDS using
// the CDNA5 async global->LDS path (ASYNCcnt-tracked, no VGPR round trip).
// 256 threads x 4 rounds x 16B = 16KB. Followed by s_wait_asynccnt 0.
__device__ __forceinline__ void async_stage_16k(const float* __restrict__ gsrc,
                                                void* ldst, int tid) {
  uint32_t l0 = (uint32_t)(uintptr_t)ldst;   // LDS offset (low 32 bits of flat)
  uint64_t g0 = (uint64_t)(uintptr_t)gsrc;
#pragma unroll
  for (int q = 0; q < 4; ++q) {
    uint32_t lo = l0 + (uint32_t)((q * 256 + tid) * 16);
    uint64_t go = g0 + (uint64_t)((q * 256 + tid) * 16);
    asm volatile("global_load_async_to_lds_b128 %0, %1, off"
                 :: "v"(lo), "v"(go) : "memory");
  }
  asm volatile("s_wait_asynccnt 0x0" ::: "memory");
}

// per-row LayerNorm stats via wave32 shuffle reduction
__device__ __forceinline__ void ln_stats(const float* __restrict__ row, int ncol,
                                         int lane, float& mean, float& rstd) {
  float s = 0.f, s2 = 0.f;
  for (int i = lane; i < ncol; i += 32) {
    float v = row[i];
    s += v; s2 += v * v;
  }
#pragma unroll
  for (int mk = 16; mk >= 1; mk >>= 1) {
    s  += __shfl_xor(s,  mk, 32);
    s2 += __shfl_xor(s2, mk, 32);
  }
  mean = s / (float)ncol;
  float var = s2 / (float)ncol - mean * mean;
  rstd = rsqrtf(var + 1e-5f);
}

// ---------------- weight repack: f32 row-major [K,N] -> f16 WMMA B-fragments ----
__global__ __launch_bounds__(256) void k_pack(const float* __restrict__ W,
                                              _Float16* __restrict__ out,
                                              int K, int N) {
  int tid = blockIdx.x * blockDim.x + threadIdx.x;
  if (tid >= K * N) return;
  int h    = tid & 15;
  int lane = (tid >> 4) & 31;
  int rest = tid >> 9;
  int NT = N >> 4;
  int nt = rest % NT;
  int kt = rest / NT;
  int base = (lane < 16) ? 0 : 8;
  int k = kt * 32 + ((h < 8) ? (base + h) : (16 + base + (h - 8)));
  int n = nt * 16 + (lane & 15);
  out[tid] = (_Float16)W[(size_t)k * N + n];
}

// ---------------- K1: pp = relu(LN(pf @ p_W + p_b)) ----------------
__global__ __launch_bounds__(256) void k_pp(
    const float* __restrict__ pf, const _Float16* __restrict__ pWp,
    const float* __restrict__ p_b, const float* __restrict__ p_g,
    const float* __restrict__ p_be, float* __restrict__ pp32,
    _Float16* __restrict__ pp16) {
  __shared__ __align__(32) unsigned char smem[16 * 256 * 4 + 16 * 256 * 2];
  float*    Af32 = (float*)smem;               // [0,16K) staging; aliased by y
  _Float16* Af16 = (_Float16*)(smem + 16384);  // [16K,24K)
  float*    y    = (float*)smem;               // [0,16K) after A consumed
  const int tid = threadIdx.x, wave = tid >> 5, lane = tid & 31;
  const int row0 = blockIdx.x * 16;
  const float* A = pf + (size_t)row0 * PD_;
  // async-stage the contiguous 16KB A tile into LDS, convert once to f16
  async_stage_16k(A, Af32, tid);
  __syncthreads();
  for (int r = 0; r < 16; ++r) Af16[r * 256 + tid] = (_Float16)Af32[r * 256 + tid];
  __syncthreads();
  v8f c0 = {}, c1 = {};
  const int nt0 = wave * 2;
#pragma unroll
  for (int kt = 0; kt < 8; ++kt) {
    v16h a  = load_a_s(Af16, 256, kt * 32, lane);
    v16h b0 = load_b(pWp, 16, kt, nt0, lane);
    v16h b1 = load_b(pWp, 16, kt, nt0 + 1, lane);
    c0 = wmma_f16(a, b0, c0);
    c1 = wmma_f16(a, b1, c1);
  }
  const int rofs = (lane < 16) ? 0 : 8, cl = lane & 15;
#pragma unroll
  for (int j = 0; j < 8; ++j) {
    int r = j + rofs;
    int col0 = nt0 * 16 + cl, col1 = col0 + 16;
    y[r * 256 + col0] = c0[j] + p_b[col0];
    y[r * 256 + col1] = c1[j] + p_b[col1];
  }
  __syncthreads();
  for (int r = wave * 2; r < wave * 2 + 2; ++r) {
    float mean, rstd;
    ln_stats(&y[r * 256], 256, lane, mean, rstd);
    for (int i = lane; i < 256; i += 32) {
      float v = (y[r * 256 + i] - mean) * rstd * p_g[i] + p_be[i];
      v = fmaxf(v, 0.f);
      pp32[(size_t)(row0 + r) * H_ + i] = v;
      pp16[(size_t)(row0 + r) * H_ + i] = (_Float16)v;
    }
  }
}

// ---------------- K2: fused vp -> cat -> h -> attn ----------------
__global__ __launch_bounds__(256) void k_attn(
    const float* __restrict__ vf, const _Float16* __restrict__ pp16,
    const _Float16* __restrict__ vWp, const float* __restrict__ v_b,
    const float* __restrict__ v_g, const float* __restrict__ v_be,
    const _Float16* __restrict__ a1p, const float* __restrict__ a1_b,
    const float* __restrict__ a_g, const float* __restrict__ a_be,
    const float* __restrict__ a2_W, const float* __restrict__ a2_b,
    float* __restrict__ attn) {
  __shared__ __align__(32) unsigned char smem[16 * 512 * 2 + 16 * 256 * 4 +
                                              16 * 256 * 2];
  _Float16* cat  = (_Float16*)smem;               // [0,16K)  16x512 f16
  float*    Af32 = (float*)(smem + 16384);        // [16K,32K) staging, -> y
  float*    y    = (float*)(smem + 16384);        // [16K,32K)
  _Float16* Af16 = (_Float16*)(smem + 32768);     // [32K,40K) 16x256 f16
  const int tid = threadIdx.x, wave = tid >> 5, lane = tid & 31;
  const int vr0 = blockIdx.x * 16;
  const int b = vr0 / (M_ * NP_);
  const int m = (vr0 / NP_) % M_;
  const int p0 = vr0 % NP_;
  const float* A = vf + (size_t)((b * M_ + m) * NP_ + p0) * VD_;
  // async-stage the contiguous 16KB vf tile, waited in helper; then pp -> cat[:,0:256]
  async_stage_16k(A, Af32, tid);
  for (int r = 0; r < 16; ++r)
    cat[r * 512 + tid] = pp16[(size_t)(b * NP_ + p0 + r) * H_ + tid];
  __syncthreads();
  // convert staged A tile to f16 once (instead of once per wave)
  for (int r = 0; r < 16; ++r) Af16[r * 256 + tid] = (_Float16)Af32[r * 256 + tid];
  __syncthreads();
  // phase A: vp_raw = vf @ v_W   (A from LDS f16, B prepacked from global)
  {
    v8f c0 = {}, c1 = {};
    const int nt0 = wave * 2;
#pragma unroll
    for (int kt = 0; kt < 8; ++kt) {
      v16h a  = load_a_s(Af16, 256, kt * 32, lane);
      v16h b0 = load_b(vWp, 16, kt, nt0, lane);
      v16h b1 = load_b(vWp, 16, kt, nt0 + 1, lane);
      c0 = wmma_f16(a, b0, c0);
      c1 = wmma_f16(a, b1, c1);
    }
    const int rofs = (lane < 16) ? 0 : 8, cl = lane & 15;
#pragma unroll
    for (int j = 0; j < 8; ++j) {
      int r = j + rofs;
      int col0 = nt0 * 16 + cl, col1 = col0 + 16;
      y[r * 256 + col0] = c0[j] + v_b[col0];   // y aliases Af32 (dead now)
      y[r * 256 + col1] = c1[j] + v_b[col1];
    }
  }
  __syncthreads();
  // LN + relu -> cat[:, 256:512] (f16)
  for (int r = wave * 2; r < wave * 2 + 2; ++r) {
    float mean, rstd;
    ln_stats(&y[r * 256], 256, lane, mean, rstd);
    for (int i = lane; i < 256; i += 32) {
      float v = (y[r * 256 + i] - mean) * rstd * v_g[i] + v_be[i];
      cat[r * 512 + 256 + i] = (_Float16)fmaxf(v, 0.f);
    }
  }
  __syncthreads();
  // phase B: h_raw = cat @ a1_W   (K = 512, A from LDS)
  {
    v8f c0 = {}, c1 = {};
    const int nt0 = wave * 2;
#pragma unroll
    for (int kt = 0; kt < 16; ++kt) {
      v16h a  = load_a_s(cat, 512, kt * 32, lane);
      v16h b0 = load_b(a1p, 16, kt, nt0, lane);
      v16h b1 = load_b(a1p, 16, kt, nt0 + 1, lane);
      c0 = wmma_f16(a, b0, c0);
      c1 = wmma_f16(a, b1, c1);
    }
    const int rofs = (lane < 16) ? 0 : 8, cl = lane & 15;
#pragma unroll
    for (int j = 0; j < 8; ++j) {
      int r = j + rofs;
      int col0 = nt0 * 16 + cl, col1 = col0 + 16;
      y[r * 256 + col0] = c0[j] + a1_b[col0];
      y[r * 256 + col1] = c1[j] + a1_b[col1];
    }
  }
  __syncthreads();
  // LN + relu in place, then attn = sigmoid(h . a2_W + a2_b)  (same-wave rows)
  for (int r = wave * 2; r < wave * 2 + 2; ++r) {
    float mean, rstd;
    ln_stats(&y[r * 256], 256, lane, mean, rstd);
    for (int i = lane; i < 256; i += 32)
      y[r * 256 + i] = fmaxf((y[r * 256 + i] - mean) * rstd * a_g[i] + a_be[i], 0.f);
    float s = 0.f;
    for (int i = lane; i < 256; i += 32) s += y[r * 256 + i] * a2_W[i];
#pragma unroll
    for (int mk = 16; mk >= 1; mk >>= 1) s += __shfl_xor(s, mk, 32);
    if (lane == 0) {
      float v = 1.f / (1.f + expf(-(s + a2_b[0])));
      attn[(size_t)(b * M_ + m) * NP_ + p0 + r] = v;
    }
  }
}

// ---------------- K3a: deterministic segment means ----------------
__global__ __launch_bounds__(256) void k_seg(
    const int* __restrict__ spid, const float* __restrict__ attn,
    const float* __restrict__ pp32, float* __restrict__ a_mean,
    float* __restrict__ f_mean) {
  const int seg = blockIdx.x;        // b*S + s
  const int b = seg / S_, s = seg % S_;
  const int t = threadIdx.x;
  const int base = b * NP_;
  float facc = 0.f, aacc = 0.f;
  int cnt = 0;
  for (int p = 0; p < NP_; ++p) {
    if (spid[base + p] == s) {
      ++cnt;
      facc += pp32[(size_t)(base + p) * H_ + t];
      if (t < M_) aacc += attn[(size_t)(b * M_ + t) * NP_ + p];
    }
  }
  const float inv = 1.f / fmaxf((float)cnt, 1.f);
  f_mean[(size_t)seg * H_ + t] = facc * inv;
  if (t < M_) a_mean[seg * M_ + t] = aacc * inv;
}

// ---------------- K3b: refinement + softmax over views ----------------
__global__ __launch_bounds__(256) void k_refine(
    const int* __restrict__ spid, const float* __restrict__ attn,
    const float* __restrict__ pp32, const float* __restrict__ a_mean,
    const float* __restrict__ f_mean, float* __restrict__ na) {
  const int wave = threadIdx.x >> 5, lane = threadIdx.x & 31;
  const int pt = blockIdx.x * 8 + wave;
  const int b = pt / NP_, p = pt % NP_;
  const int sp = spid[pt];
  const bool valid = sp >= 0;
  const int seg = valid ? (b * S_ + sp) : 0;
  float dot = 0.f, nf = 0.f, nm = 0.f;
  for (int i = lane; i < 256; i += 32) {
    float fv  = pp32[(size_t)pt * H_ + i];
    float fmv = f_mean[(size_t)seg * H_ + i];
    dot += fv * fmv; nf += fv * fv; nm += fmv * fmv;
  }
#pragma unroll
  for (int mk = 16; mk >= 1; mk >>= 1) {
    dot += __shfl_xor(dot, mk, 32);
    nf  += __shfl_xor(nf,  mk, 32);
    nm  += __shfl_xor(nm,  mk, 32);
  }
  const float sim =
      dot / (fmaxf(sqrtf(nf), 1e-8f) * fmaxf(sqrtf(nm), 1e-8f));
  float ref = 0.f;
  if (lane < M_) {
    float a  = attn[(size_t)(b * M_ + lane) * NP_ + p];
    float am = a_mean[seg * M_ + lane];
    ref = valid ? (am + (a - am) * sim) : a;
  }
  float mx = ref;
#pragma unroll
  for (int mk = 4; mk >= 1; mk >>= 1) mx = fmaxf(mx, __shfl_xor(mx, mk, 8));
  float e = (lane < M_) ? expf(ref - mx) : 0.f;
  float ss = e;
#pragma unroll
  for (int mk = 4; mk >= 1; mk >>= 1) ss += __shfl_xor(ss, mk, 8);
  if (lane < M_) na[(size_t)pt * M_ + lane] = e / ss;
}

// ---------------- K4: weighted sum -> wvp -> Z_PV ----------------
__global__ __launch_bounds__(256) void k_out(
    const float* __restrict__ vf, const _Float16* __restrict__ pp16,
    const float* __restrict__ na, const _Float16* __restrict__ vWp,
    const float* __restrict__ v_b, const float* __restrict__ v_g,
    const float* __restrict__ v_be, const _Float16* __restrict__ fWp,
    const float* __restrict__ f_b, const float* __restrict__ f_g,
    const float* __restrict__ f_be, float* __restrict__ out) {
  __shared__ __align__(32) unsigned char smem[49152];
  _Float16* cat = (_Float16*)smem;               // [0,16K):   16x512 f16
  float* y      = (float*)(smem + 16384);        // [16K,32K): 16x256 f32 (phase B)
  _Float16* wgt = (_Float16*)(smem + 32768);     // [32K,40K): 16x256 f16
  float* na_s   = (float*)(smem + 40960);        // [40K,40.5K): 16x8
  float* yy     = (float*)(smem + 16384);        // [16K,48K): 16x512 f32 (phase C)
  const int tid = threadIdx.x, wave = tid >> 5, lane = tid & 31;
  const int bp0 = blockIdx.x * 16;
  const int b = bp0 / NP_, p0 = bp0 % NP_;
  if (tid < 128) na_s[tid] = na[(size_t)(bp0 + (tid >> 3)) * M_ + (tid & 7)];
  for (int r = 0; r < 16; ++r)
    cat[r * 512 + tid] = pp16[(size_t)(bp0 + r) * H_ + tid];
  __syncthreads();
  // weighted[r, tid] = sum_m na[r,m] * vf[b,m,p0+r,tid]
  for (int r = 0; r < 16; ++r) {
    float acc = 0.f;
#pragma unroll
    for (int m = 0; m < M_; ++m)
      acc += na_s[r * 8 + m] *
             vf[(size_t)((b * M_ + m) * NP_ + p0 + r) * VD_ + tid];
    wgt[r * 256 + tid] = (_Float16)acc;
  }
  __syncthreads();
  // phase B: wvp_raw = weighted @ v_W
  {
    v8f c0 = {}, c1 = {};
    const int nt0 = wave * 2;
#pragma unroll
    for (int kt = 0; kt < 8; ++kt) {
      v16h a  = load_a_s(wgt, 256, kt * 32, lane);
      v16h b0 = load_b(vWp, 16, kt, nt0, lane);
      v16h b1 = load_b(vWp, 16, kt, nt0 + 1, lane);
      c0 = wmma_f16(a, b0, c0);
      c1 = wmma_f16(a, b1, c1);
    }
    const int rofs = (lane < 16) ? 0 : 8, cl = lane & 15;
#pragma unroll
    for (int j = 0; j < 8; ++j) {
      int r = j + rofs;
      int col0 = nt0 * 16 + cl, col1 = col0 + 16;
      y[r * 256 + col0] = c0[j] + v_b[col0];
      y[r * 256 + col1] = c1[j] + v_b[col1];
    }
  }
  __syncthreads();
  // LN + relu -> cat[:, 256:512]
  for (int r = wave * 2; r < wave * 2 + 2; ++r) {
    float mean, rstd;
    ln_stats(&y[r * 256], 256, lane, mean, rstd);
    for (int i = lane; i < 256; i += 32) {
      float v = (y[r * 256 + i] - mean) * rstd * v_g[i] + v_be[i];
      cat[r * 512 + 256 + i] = (_Float16)fmaxf(v, 0.f);
    }
  }
  __syncthreads();
  // phase C: Z_raw = cat @ f_W   (K=512, N=512: 32 tiles, 4 per wave)
  {
    v8f z = {};
    v8f c[4] = {z, z, z, z};
    const int nt0 = wave * 4;
#pragma unroll
    for (int kt = 0; kt < 16; ++kt) {
      v16h a = load_a_s(cat, 512, kt * 32, lane);
#pragma unroll
      for (int t = 0; t < 4; ++t) {
        v16h bb = load_b(fWp, 32, kt, nt0 + t, lane);
        c[t] = wmma_f16(a, bb, c[t]);
      }
    }
    const int rofs = (lane < 16) ? 0 : 8, cl = lane & 15;
#pragma unroll
    for (int t = 0; t < 4; ++t) {
      int col = (nt0 + t) * 16 + cl;
#pragma unroll
      for (int j = 0; j < 8; ++j) {
        int r = j + rofs;
        yy[r * 512 + col] = c[t][j] + f_b[col];
      }
    }
  }
  __syncthreads();
  // final LayerNorm (no relu) -> output
  for (int r = wave * 2; r < wave * 2 + 2; ++r) {
    float mean, rstd;
    ln_stats(&yy[r * 512], 512, lane, mean, rstd);
    for (int i = lane; i < 512; i += 32)
      out[(size_t)(bp0 + r) * FD_ + i] =
          (yy[r * 512 + i] - mean) * rstd * f_g[i] + f_be[i];
  }
}

// ---------------- host launcher ----------------
extern "C" void kernel_launch(void* const* d_in, const int* in_sizes, int n_in,
                              void* d_out, int out_size, void* d_ws,
                              size_t ws_size, hipStream_t stream) {
  (void)in_sizes; (void)n_in; (void)out_size; (void)ws_size;
  const float* pf   = (const float*)d_in[0];
  const float* vf   = (const float*)d_in[1];
  const int*   spid = (const int*)d_in[2];
  const float* p_W  = (const float*)d_in[3];
  const float* p_b  = (const float*)d_in[4];
  const float* p_g  = (const float*)d_in[5];
  const float* p_be = (const float*)d_in[6];
  const float* v_W  = (const float*)d_in[7];
  const float* v_b  = (const float*)d_in[8];
  const float* v_g  = (const float*)d_in[9];
  const float* v_be = (const float*)d_in[10];
  const float* a1_W = (const float*)d_in[11];
  const float* a1_b = (const float*)d_in[12];
  const float* a_g  = (const float*)d_in[13];
  const float* a_be = (const float*)d_in[14];
  const float* a2_W = (const float*)d_in[15];
  const float* a2_b = (const float*)d_in[16];
  const float* f_W  = (const float*)d_in[17];
  const float* f_b  = (const float*)d_in[18];
  const float* f_g  = (const float*)d_in[19];
  const float* f_be = (const float*)d_in[20];
  float* out = (float*)d_out;

  char* ws = (char*)d_ws;
  _Float16* pWp  = (_Float16*)(ws + 0);                    // 256*256*2 = 128K
  _Float16* vWp  = (_Float16*)(ws + (131072));             // 128K
  _Float16* a1p  = (_Float16*)(ws + (262144));             // 512*256*2 = 256K
  _Float16* fWp  = (_Float16*)(ws + (524288));             // 512*512*2 = 512K
  float*    pp32 = (float*)(ws + (1048576));               // 16384*256*4 = 16M
  _Float16* pp16 = (_Float16*)(ws + (17825792));           // 8M
  float*    attn = (float*)(ws + (26214400));              // 131072*4 = 512K
  float*    amn  = (float*)(ws + (26738688));              // 512*8*4 = 16K
  float*    fmn  = (float*)(ws + (26755072));              // 512*256*4 = 512K
  float*    na   = (float*)(ws + (27279360));              // 131072*4 = 512K

  // 1) repack weights into WMMA B-fragment layout (f16)
  k_pack<<<(256 * 256) / 256, 256, 0, stream>>>(p_W, pWp, 256, 256);
  k_pack<<<(256 * 256) / 256, 256, 0, stream>>>(v_W, vWp, 256, 256);
  k_pack<<<(512 * 256) / 256, 256, 0, stream>>>(a1_W, a1p, 512, 256);
  k_pack<<<(512 * 512) / 256, 256, 0, stream>>>(f_W, fWp, 512, 512);

  // 2) pp
  k_pp<<<(B_ * NP_) / 16, 256, 0, stream>>>(pf, pWp, p_b, p_g, p_be, pp32, pp16);

  // 3) fused vp/cat/h/attn
  k_attn<<<(B_ * M_ * NP_) / 16, 256, 0, stream>>>(
      vf, pp16, vWp, v_b, v_g, v_be, a1p, a1_b, a_g, a_be, a2_W, a2_b, attn);

  // 4) segment means (deterministic)
  k_seg<<<B_ * S_, 256, 0, stream>>>(spid, attn, pp32, amn, fmn);

  // 5) refinement + softmax over views
  k_refine<<<(B_ * NP_) / 8, 256, 0, stream>>>(spid, attn, pp32, amn, fmn, na);

  // 6) weighted fuse -> output
  k_out<<<(B_ * NP_) / 16, 256, 0, stream>>>(
      vf, pp16, na, vWp, v_b, v_g, v_be, fWp, f_b, f_g, f_be, out);
}
